// MetaNetImageEncoder_68504728371608
// MI455X (gfx1250) — matrix operand order repaired and checked by reading, not compiled, not run on hardware
//
#include <hip/hip_runtime.h>
#include <hip/hip_bf16.h>

typedef __attribute__((ext_vector_type(16))) __bf16 v16bf;
typedef __attribute__((ext_vector_type(8)))  __bf16 v8bf;
typedef __attribute__((ext_vector_type(8)))  float  v8f;

#define BATCH 32
#define MPAD  208     // 196 patches padded to 13*16
#define ROWS_VALID 196

// ---------------------------------------------------------------------------
// Patchify: x[B,3,224,224] -> P[B,208,768] fp32 (rows >=196 zero)
// p[b, py*14+px, c*256+iy*16+ix] = x[b,c,py*16+iy,px*16+ix]
// ---------------------------------------------------------------------------
__global__ __launch_bounds__(256) void patchify_kernel(const float* __restrict__ x,
                                                       float* __restrict__ P) {
  int idx = blockIdx.x * 256 + threadIdx.x;          // 32*208*768 exact
  int d = idx % 768;
  int r = (idx / 768) % MPAD;
  int b = idx / (768 * MPAD);
  float v = 0.f;
  if (r < ROWS_VALID) {
    int py = r / 14, px = r % 14;
    int c = d >> 8, rem = d & 255, iy = rem >> 4, ix = rem & 15;
    v = x[(((size_t)(b * 3 + c) * 224) + py * 16 + iy) * 224 + px * 16 + ix];
  }
  P[idx] = v;
}

// ---------------------------------------------------------------------------
// WMMA GEMM:  C[b] = act( A[b] (MxK) * Bw (KxN) + bias )
//   COMPOSE:  Bw = B0 + sum_t coefs[b,t]*dB[t],  bias = bias0 + sum_t c*dbias[t]
//   POOL:     no C store; per-(b, m-tile) column partial sums of relu(out)
//             for rows < rowsValid into pool[b][slot][N]  (deterministic)
// Block: 256 thr = 8 waves (2M x 4N), wave tile 16x64, block tile 32x256, K-step 32.
// A-row OOB handling is branch-free: clamped row + 0/1 float mask (M >= 32 always).
// ---------------------------------------------------------------------------
template <bool COMPOSE, bool POOL, bool RELU>
__global__ __launch_bounds__(256)
void gemm_wmma(const float* __restrict__ A, long long strideAb, int M, int N, int K,
               const float* __restrict__ B0, const float* __restrict__ dB,
               const float* __restrict__ coefs,
               const float* __restrict__ bias, const float* __restrict__ dbias,
               float* __restrict__ C, long long strideCb,
               float* __restrict__ pool, int rowsValid) {
  __shared__ __bf16 As[32 * 40];    // A tile 32(M) x 32(K), row pitch 40 (80B)
  __shared__ __bf16 Bs[256 * 40];   // B tile transposed: 256(N) x 32(K), pitch 40

  const int tid  = threadIdx.x;
  const int b    = blockIdx.z;
  const float* Ab = A + (size_t)b * strideAb;

  float cc[8];
  if (COMPOSE) {
#pragma unroll
    for (int t = 0; t < 8; ++t) cc[t] = coefs[b * 8 + t];
  }

  const int wave = tid >> 5, lane = tid & 31;
  const int wm = wave >> 2, wn = wave & 3;        // 2 x 4 wave grid
  const int g = lane >> 4, ln = lane & 15;
  const int m0 = blockIdx.y * 32 + wm * 16;
  const int n0 = blockIdx.x * 256 + wn * 64;

  v8f acc[4];
#pragma unroll
  for (int s = 0; s < 4; ++s)
#pragma unroll
    for (int r = 0; r < 8; ++r) acc[s][r] = 0.f;

  // staging decomposition
  const int a_m  = tid >> 4;         // 0..15
  const int a_k2 = (tid & 15) * 2;   // even k
  const int b_k0 = (tid >> 6) * 8;   // 0,8,16,24
  const int b_n  = tid & 63;         // 0..63

  // Branch-free A staging setup (loop-invariant): clamp row, 0/1 mask.
  const float* aRowPtr[2];
  float aMask[2];
#pragma unroll
  for (int mi = 0; mi < 2; ++mi) {
    int row = blockIdx.y * 32 + a_m + 16 * mi;
    int rowc = row < M ? row : (M - 1);           // always a valid row (M >= 32)
    aRowPtr[mi] = Ab + (size_t)rowc * K + a_k2;
    aMask[mi]   = row < M ? 1.f : 0.f;
  }

  for (int kb = 0; kb < K; kb += 32) {
    // ---- stage A (fp32 -> bf16), masked, straight-line
#pragma unroll
    for (int mi = 0; mi < 2; ++mi) {
      int m = a_m + 16 * mi;
      float f0 = aRowPtr[mi][kb]     * aMask[mi];
      float f1 = aRowPtr[mi][kb + 1] * aMask[mi];
      __bf16* dst = &As[m * 40 + a_k2];
      dst[0] = (__bf16)f0; dst[1] = (__bf16)f1;
    }
    // ---- stage B transposed (compose on the fly, fp32 -> bf16)
#pragma unroll
    for (int j = 0; j < 4; ++j) {
      int n = blockIdx.x * 256 + b_n + 64 * j;
      __bf16 vals[8];
#pragma unroll
      for (int i = 0; i < 8; ++i) {
        int k = kb + b_k0 + i;
        float v = B0[(size_t)k * N + n];
        if (COMPOSE) {
#pragma unroll
          for (int t = 0; t < 8; ++t)
            v += cc[t] * dB[((size_t)t * K + k) * N + n];
        }
        vals[i] = (__bf16)v;
      }
      __bf16* dst = &Bs[(b_n + 64 * j) * 40 + b_k0];
#pragma unroll
      for (int i = 0; i < 8; ++i) dst[i] = vals[i];
    }
    __syncthreads();

    // ---- compute: 1 A frag + 4 B frags -> 4 WMMAs
    {
      const __bf16* ap = &As[(wm * 16 + ln) * 40 + 8 * g];
      v8bf alo = *(const v8bf*)ap;           // K = 8g .. 8g+7
      v8bf ahi = *(const v8bf*)(ap + 16);    // K = 16+8g .. 16+8g+7
      v16bf a = __builtin_shufflevector(alo, ahi, 0, 1, 2, 3, 4, 5, 6, 7,
                                        8, 9, 10, 11, 12, 13, 14, 15);
#pragma unroll
      for (int s = 0; s < 4; ++s) {
        const __bf16* bp = &Bs[(wn * 64 + s * 16 + ln) * 40 + 8 * g];
        v8bf blo = *(const v8bf*)bp;         // K = 8g .. 8g+7
        v8bf bhi = *(const v8bf*)(bp + 16);  // K = 16+8g .. 16+8g+7
        v16bf bb = __builtin_shufflevector(blo, bhi, 0, 1, 2, 3, 4, 5, 6, 7,
                                           8, 9, 10, 11, 12, 13, 14, 15);
        acc[s] = __builtin_amdgcn_wmma_f32_16x16x32_bf16(
            false, a, false, bb, (short)0, acc[s], false, false);
      }
    }
    __syncthreads();
  }

  // ---- epilogue
#pragma unroll
  for (int s = 0; s < 4; ++s) {
    int col = n0 + s * 16 + ln;
    float bv = 0.f;
    if (bias != nullptr) {
      bv = bias[col];
      if (COMPOSE) {
#pragma unroll
        for (int t = 0; t < 8; ++t) bv += cc[t] * dbias[t * N + col];
      }
    }
    if (POOL) {
      float ps = 0.f;
#pragma unroll
      for (int r = 0; r < 8; ++r) {
        int row = m0 + 8 * g + r;
        if (row < rowsValid) {
          float v = acc[s][r] + bv;
          if (RELU) v = v > 0.f ? v : 0.f;
          ps += v;
        }
      }
      ps += __shfl_xor(ps, 16);                 // combine both row-halves
      if (g == 0) {
        int slot = blockIdx.y * 2 + wm;         // 0..13
        pool[((size_t)b * 14 + slot) * N + col] = ps;
      }
    } else {
      float* Cb = C + (size_t)b * strideCb;
#pragma unroll
      for (int r = 0; r < 8; ++r) {
        int row = m0 + 8 * g + r;
        if (row < M) {
          float v = acc[s][r] + bv;
          if (RELU) v = v > 0.f ? v : 0.f;
          Cb[(size_t)row * N + col] = v;
        }
      }
    }
  }
}

// pool[b][n] = scale * sum_{slot<14} partial[b][slot][n]
__global__ __launch_bounds__(256) void pool_reduce(const float* __restrict__ partial,
                                                   float* __restrict__ outp,
                                                   int N, float scale) {
  int idx = blockIdx.x * 256 + threadIdx.x;   // 32*N exact
  int b = idx / N, n = idx % N;
  float s = 0.f;
  for (int t = 0; t < 14; ++t) s += partial[((size_t)b * 14 + t) * N + n];
  outp[idx] = s * scale;
}

// m1 = relu(base @ mW1 + mb1)   [32,128], K=512
__global__ __launch_bounds__(256) void meta1_kernel(const float* __restrict__ base,
                                                    const float* __restrict__ mW1,
                                                    const float* __restrict__ mb1,
                                                    float* __restrict__ m1) {
  int idx = blockIdx.x * 256 + threadIdx.x;   // 4096
  int b = idx >> 7, h = idx & 127;
  float acc = mb1[h];
  for (int k = 0; k < 512; ++k) acc += base[b * 512 + k] * mW1[k * 128 + h];
  m1[idx] = acc > 0.f ? acc : 0.f;
}

// coefs = m1 @ mW2 + mb2   [32,8], K=128
__global__ __launch_bounds__(256) void meta2_kernel(const float* __restrict__ m1,
                                                    const float* __restrict__ mW2,
                                                    const float* __restrict__ mb2,
                                                    float* __restrict__ coefs) {
  int idx = threadIdx.x;                      // 256
  int b = idx >> 3, t = idx & 7;
  float acc = mb2[t];
  for (int k = 0; k < 128; ++k) acc += m1[b * 128 + k] * mW2[k * 8 + t];
  coefs[idx] = acc;
}

// out[b,f] = q[8][b,f] + b3[f] + sum_t c[b,t]*(q[t][b,f] + db3[t,f])
__global__ __launch_bounds__(256) void finalize_kernel(const float* __restrict__ q,
                                                       const float* __restrict__ coefs,
                                                       const float* __restrict__ b3,
                                                       const float* __restrict__ db3,
                                                       float* __restrict__ out) {
  int idx = blockIdx.x * 256 + threadIdx.x;   // 32*512 exact
  int b = idx >> 9, f = idx & 511;
  float v = q[(size_t)8 * BATCH * 512 + idx] + b3[f];
#pragma unroll
  for (int t = 0; t < 8; ++t)
    v += coefs[b * 8 + t] * (q[(size_t)t * BATCH * 512 + idx] + db3[t * 512 + f]);
  out[idx] = v;
}

extern "C" void kernel_launch(void* const* d_in, const int* in_sizes, int n_in,
                              void* d_out, int out_size, void* d_ws, size_t ws_size,
                              hipStream_t stream) {
  const float* x   = (const float*)d_in[0];
  const float* W1  = (const float*)d_in[1];
  const float* b1  = (const float*)d_in[2];
  const float* W2  = (const float*)d_in[3];
  const float* b2  = (const float*)d_in[4];
  const float* W3  = (const float*)d_in[5];
  const float* b3  = (const float*)d_in[6];
  const float* dW1 = (const float*)d_in[7];
  const float* db1 = (const float*)d_in[8];
  const float* dW2 = (const float*)d_in[9];
  const float* db2 = (const float*)d_in[10];
  const float* dW3 = (const float*)d_in[11];
  const float* db3 = (const float*)d_in[12];
  const float* mW1 = (const float*)d_in[13];
  const float* mb1 = (const float*)d_in[14];
  const float* mW2 = (const float*)d_in[15];
  const float* mb2 = (const float*)d_in[16];

  // workspace carve (floats); every buffer fully written before read each call
  float* ws      = (float*)d_ws;
  float* P       = ws;                                  // 32*208*768
  float* Hbase   = P       + (size_t)BATCH * MPAD * 768;
  float* Hpers   = Hbase   + (size_t)BATCH * MPAD * 768;
  float* partial = Hpers   + (size_t)BATCH * MPAD * 768; // 32*14*3072
  float* pool1   = partial + (size_t)BATCH * 14 * 3072;  // 32*3072
  float* pool2   = pool1   + (size_t)BATCH * 3072;
  float* base    = pool2   + (size_t)BATCH * 3072;       // 32*512
  float* m1      = base    + (size_t)BATCH * 512;        // 32*128
  float* coefs   = m1      + (size_t)BATCH * 128;        // 32*8
  float* q       = coefs   + (size_t)BATCH * 8;          // 9*32*512

  const long long sA = (long long)MPAD * 768;

  // 0) patchify
  patchify_kernel<<<(BATCH * MPAD * 768) / 256, 256, 0, stream>>>(x, P);

  // 1) base: Hbase = relu(P @ W1 + b1)           M=208,K=768,N=768 per b
  gemm_wmma<false, false, true><<<dim3(3, 7, BATCH), 256, 0, stream>>>(
      P, sA, MPAD, 768, 768, W1, nullptr, nullptr, b1, nullptr, Hbase, sA, nullptr, 0);

  // 2) base: pooled relu(Hbase @ W2 + b2)        -> partial -> pool1 (mean/196)
  gemm_wmma<false, true, true><<<dim3(12, 7, BATCH), 256, 0, stream>>>(
      Hbase, sA, MPAD, 3072, 768, W2, nullptr, nullptr, b2, nullptr,
      nullptr, 0, partial, ROWS_VALID);
  pool_reduce<<<(BATCH * 3072) / 256, 256, 0, stream>>>(partial, pool1, 3072, 1.f / 196.f);

  // 3) base = pool1 @ W3 + b3                    M=32,K=3072,N=512
  gemm_wmma<false, false, false><<<dim3(2, 1, 1), 256, 0, stream>>>(
      pool1, 0, 32, 512, 3072, W3, nullptr, nullptr, b3, nullptr, base, 0, nullptr, 0);

  // 4) meta-net -> coefs[32,8]
  meta1_kernel<<<16, 256, 0, stream>>>(base, mW1, mb1, m1);
  meta2_kernel<<<1, 256, 0, stream>>>(m1, mW2, mb2, coefs);

  // 5) personalized: Hpers = relu(P @ (W1 + sum c dW1) + (b1 + c@db1))
  gemm_wmma<true, false, true><<<dim3(3, 7, BATCH), 256, 0, stream>>>(
      P, sA, MPAD, 768, 768, W1, dW1, coefs, b1, db1, Hpers, sA, nullptr, 0);

  // 6) personalized pooled layer 2 -> pool2
  gemm_wmma<true, true, true><<<dim3(12, 7, BATCH), 256, 0, stream>>>(
      Hpers, sA, MPAD, 3072, 768, W2, dW2, coefs, b2, db2,
      nullptr, 0, partial, ROWS_VALID);
  pool_reduce<<<(BATCH * 3072) / 256, 256, 0, stream>>>(partial, pool2, 3072, 1.f / 196.f);

  // 7) q[t] = pool2 @ dW3[t]  (t<8),  q[8] = pool2 @ W3
  for (int t = 0; t < 9; ++t) {
    const float* Bp = (t < 8) ? (dW3 + (size_t)t * 3072 * 512) : W3;
    gemm_wmma<false, false, false><<<dim3(2, 1, 1), 256, 0, stream>>>(
        pool2, 0, 32, 512, 3072, Bp, nullptr, nullptr, nullptr, nullptr,
        q + (size_t)t * BATCH * 512, 0, nullptr, 0);
  }

  // 8) out = q[8] + b3 + sum_t c_t*(q[t] + db3[t])
  finalize_kernel<<<(BATCH * 512) / 256, 256, 0, stream>>>(q, coefs, b3, db3, (float*)d_out);
}
